// GCN_66425964200658
// MI455X (gfx1250) — matrix operand order repaired and checked by reading, not compiled, not run on hardware
//
#include <hip/hip_runtime.h>
#include <math.h>

static constexpr int kN = 384, kD = 256, kR = 64, kRelH = 128, kNeiH = 256;

typedef _Float16 h16 __attribute__((ext_vector_type(16)));
typedef _Float16 h8  __attribute__((ext_vector_type(8)));
typedef float    f8  __attribute__((ext_vector_type(8)));
typedef unsigned int u32x4 __attribute__((ext_vector_type(4)));
typedef int          i32x4 __attribute__((ext_vector_type(4)));
typedef int          i32x8 __attribute__((ext_vector_type(8)));

__device__ __forceinline__ f8 wmma_f16f32(h16 a, h16 b, f8 c) {
  // 8 args: (neg_a, A, neg_b, B, c_mod, C, reuse_a, reuse_b)
  return __builtin_amdgcn_wmma_f32_16x16x32_f16(false, a, false, b, (short)0, c, false, false);
}

// ---------------------------------------------------------------------------
// TDM: DMA a contiguous run of `nhalf` f16 elements from global -> LDS.
// Encodes ONE DMA op per issuing wave; caller gates to a single wave and the
// helper drains TENSORcnt; caller publishes via __syncthreads().
// D# layout per CDNA5 ISA ch.8: group0 {count,lds,global,type=2},
// group1 {data_size=2B, tensor_dim0=tile_dim0=nhalf, tensor_dim1=1, stride}.
// This toolchain's builtin takes 6 args (groups 0..3 + extra group + cpol).
// ---------------------------------------------------------------------------
#if __has_builtin(__builtin_amdgcn_tensor_load_to_lds)
#define HAVE_TDM 1
#else
#define HAVE_TDM 0
#endif

__device__ __forceinline__ void tdm_load_halves(void* lds_dst, const void* gsrc,
                                                unsigned nhalf) {
#if HAVE_TDM
  unsigned long long ga = (unsigned long long)gsrc;
  unsigned lds = (unsigned)(size_t)lds_dst;   // LDS aperture: addr[31:0] = LDS offset
  u32x4 g0;
  g0[0] = 1u;                                           // count=1 (valid user D#)
  g0[1] = lds;                                          // lds_addr
  g0[2] = (unsigned)ga;                                 // global_addr[31:0]
  g0[3] = (unsigned)((ga >> 32) & 0x01FFFFFFu) | (2u << 30);  // addr[56:32] | type=2
  i32x8 g1;
  g1[0] = 0x00010000;                                   // data_size=1 -> 2 bytes
  g1[1] = (int)((nhalf & 0xFFFFu) << 16);               // tensor_dim0[15:0]
  g1[2] = (int)((nhalf >> 16) | (1u << 16));            // tensor_dim0[31:16] | tensor_dim1=1
  g1[3] = (int)((nhalf & 0xFFFFu) << 16);               // tile_dim0 = nhalf
  g1[4] = 0;                                            // tile_dim1=0, tile_dim2=0
  g1[5] = (int)nhalf;                                   // tensor_dim0_stride[31:0]
  g1[6] = 0;
  g1[7] = 0;
  i32x4 z4 = {0, 0, 0, 0};
  i32x8 z8 = {0, 0, 0, 0, 0, 0, 0, 0};
  __builtin_amdgcn_tensor_load_to_lds(g0, g1, z4, z4, z8, 0);
  __builtin_amdgcn_s_wait_tensorcnt(0);
#else
  (void)lds_dst; (void)gsrc; (void)nhalf;
#endif
}

// ---------------------------------------------------------------------------
// Prep 1: transpose+convert weight slices to f16 for WMMA B operands.
//   wrT[d][k] = ngate_w[k][d]   (first 64 rows of ngate_w)   -> [256][64]
//   w2T[n][k] = rel_w2[k][n]                                 -> [64][128]
// ---------------------------------------------------------------------------
__global__ void k_prep_convert(const float* __restrict__ ngate_w,
                               const float* __restrict__ rel_w2,
                               _Float16* __restrict__ wrT,
                               _Float16* __restrict__ w2T) {
  int t = blockIdx.x * blockDim.x + threadIdx.x;
  if (t < kD * kR) {                    // 16384
    int d = t >> 6, k = t & 63;
    wrT[t] = (_Float16)ngate_w[k * kD + d];
  }
  if (t < kR * kRelH) {                 // 8192
    int n = t >> 7, k = t & 127;
    w2T[t] = (_Float16)rel_w2[k * kR + n];
  }
}

// ---------------------------------------------------------------------------
// Prep 2 (per row i): P1[i] = h_i@Wh1 + ngate_b, P2[i] = h_i@Wh2,
//                     q1[i] = h_i . war_w[64:320] + war_b, q2[i] = h_i . war_w[320:576]
// ---------------------------------------------------------------------------
__global__ void k_prep_rows(const float* __restrict__ self_h,
                            const float* __restrict__ ngate_w,
                            const float* __restrict__ ngate_b,
                            const float* __restrict__ war_w,
                            const float* __restrict__ war_b,
                            float* __restrict__ P1, float* __restrict__ P2,
                            float* __restrict__ q1, float* __restrict__ q2) {
  int i = blockIdx.x, t = threadIdx.x;
  __shared__ float hrow[kD];
  __shared__ float red1[256], red2[256];
  hrow[t] = self_h[i * kD + t];
  __syncthreads();
  float a1 = 0.f, a2 = 0.f;
  for (int k = 0; k < kD; ++k) {
    float hv = hrow[k];
    a1 = fmaf(hv, ngate_w[(kR + k) * kD + t], a1);
    a2 = fmaf(hv, ngate_w[(kR + kD + k) * kD + t], a2);
  }
  P1[i * kD + t] = a1 + ngate_b[t];
  P2[i * kD + t] = a2;
  red1[t] = hrow[t] * war_w[kR + t];
  red2[t] = hrow[t] * war_w[kR + kD + t];
  __syncthreads();
  for (int s = 128; s > 0; s >>= 1) {
    if (t < s) { red1[t] += red1[t + s]; red2[t] += red2[t + s]; }
    __syncthreads();
  }
  if (t == 0) { q1[i] = red1[0] + war_b[0]; q2[i] = red2[0]; }
}

// ---------------------------------------------------------------------------
// Pairs kernel: per pair p=(i,j), compute
//   h1 = relu(corr @ rel_w1 + b1)              (K=2, VALU)
//   r  = relu(h1 @ rel_w2 + b2)                (WMMA 16x16x32 f16, K=128, N=64)
//   tt = r . war_w[0:64] + q1[i] + q2[j]       (fused in accumulator epilogue)
// Block = 256 thr = 8 waves; wave tile = 16 pairs; block = 128 pairs.
// ---------------------------------------------------------------------------
__global__ void k_pairs(const float* __restrict__ corr,
                        const float* __restrict__ rel_w1,
                        const float* __restrict__ rel_b1,
                        const float* __restrict__ rel_b2,
                        const _Float16* __restrict__ w2T,
                        const float* __restrict__ war_w,
                        const float* __restrict__ q1,
                        const float* __restrict__ q2,
                        _Float16* __restrict__ r16,
                        float* __restrict__ tt) {
  __shared__ alignas(32) _Float16 h1s[8][16][kRelH];   // 32 KB
  __shared__ alignas(32) _Float16 w2Ts[kR][kRelH];     // 16 KB
  __shared__ float w1s[2][kRelH];
  __shared__ float b1s[kRelH];
  __shared__ float b2s[kR];
  __shared__ float warrS[kR];
  int tid = threadIdx.x;
  int wave = tid >> 5, lane = tid & 31;
  int lo = lane & 15, hi = lane >> 4, b8 = hi * 8;
  int p0 = blockIdx.x * 128 + wave * 16;

#if HAVE_TDM
  if (wave == 0) tdm_load_halves(&w2Ts[0][0], w2T, kR * kRelH);  // 16 KB via TDM
#else
  {
    const uint4* src = (const uint4*)w2T;
    uint4* dst = (uint4*)&w2Ts[0][0];
    for (int idx = tid; idx < 1024; idx += 256) dst[idx] = src[idx];
  }
#endif
  if (tid < kRelH) {
    w1s[0][tid] = rel_w1[tid];
    w1s[1][tid] = rel_w1[kRelH + tid];
    b1s[tid] = rel_b1[tid];
  }
  if (tid < kR) { b2s[tid] = rel_b2[tid]; warrS[tid] = war_w[tid]; }
  __syncthreads();

  // layer 1: 16 pairs x 128 hidden per wave
  for (int pr = 0; pr < 16; ++pr) {
    int p = p0 + pr;
    float c0 = corr[p * 2], c1 = corr[p * 2 + 1];
    for (int hu = lane; hu < kRelH; hu += 32) {
      float v = fmaf(c0, w1s[0][hu], fmaf(c1, w1s[1][hu], b1s[hu]));
      h1s[wave][pr][hu] = (_Float16)fmaxf(v, 0.f);
    }
  }
  __syncthreads();

  // layer 2: M=16 pairs, K=128 (4 chunks of 32), N=64 (4 tiles of 16)
  h16 af[4];
  for (int kc = 0; kc < 4; ++kc) {
    const h8* pa = (const h8*)&h1s[wave][lo][kc * 32 + hi * 16];
    h8 alo = pa[0], ahi = pa[1];   // K = hi*16+0..7 and K = hi*16+8..15 (+kc*32)
    for (int e = 0; e < 8; ++e) { af[kc][e] = alo[e]; af[kc][8 + e] = ahi[e]; }
  }
  float ttacc[8] = {0, 0, 0, 0, 0, 0, 0, 0};
  for (int nt = 0; nt < 4; ++nt) {
    f8 acc = {};
    for (int kc = 0; kc < 4; ++kc) {
      h16 bf = *(const h16*)&w2Ts[nt * 16 + lo][kc * 32 + hi * 16];
      acc = wmma_f16f32(af[kc], bf, acc);
    }
    int n = nt * 16 + lo;
    float bb = b2s[n], wr = warrS[n];
    for (int v = 0; v < 8; ++v) {
      float rv = fmaxf(acc[v] + bb, 0.f);         // row m = b8+v, col n
      r16[(size_t)(p0 + b8 + v) * kR + n] = (_Float16)rv;
      ttacc[v] = fmaf(rv, wr, ttacc[v]);
    }
  }
  // reduce tt over the 16 column-lanes of each half-wave
  for (int v = 0; v < 8; ++v) {
    float s = ttacc[v];
    s += __shfl_xor(s, 1, 32);
    s += __shfl_xor(s, 2, 32);
    s += __shfl_xor(s, 4, 32);
    s += __shfl_xor(s, 8, 32);
    if (lo == 0) {
      int pair = p0 + b8 + v;
      int i = pair / kN, j = pair - i * kN;
      tt[pair] = s + q1[i] + q2[j];
    }
  }
}

// ---------------------------------------------------------------------------
// Masked row softmax, replicating the reference's pos==0 -> -inf quirk.
// ---------------------------------------------------------------------------
__global__ void k_softmax(const float* __restrict__ tt,
                          const int* __restrict__ nei,
                          float* __restrict__ wgt) {
  int i = blockIdx.x, t = threadIdx.x;
  __shared__ float red[256];
  float p[2]; int msk[2];
  float mx = -INFINITY;
  for (int u = 0; u < 2; ++u) {
    int j = t + u * 256;
    if (j < kN) {
      msk[u] = nei[i * kN + j] > 0;
      float v = msk[u] ? tt[i * kN + j] : 0.f;
      v = (v == 0.f) ? -INFINITY : v;
      p[u] = v;
      mx = fmaxf(mx, v);
    } else { p[u] = -INFINITY; msk[u] = 0; }
  }
  red[t] = mx; __syncthreads();
  for (int s = 128; s > 0; s >>= 1) {
    if (t < s) red[t] = fmaxf(red[t], red[t + s]);
    __syncthreads();
  }
  float rowmax = red[0];
  __syncthreads();
  float e[2]; float lsum = 0.f;
  for (int u = 0; u < 2; ++u) {
    e[u] = (p[u] == -INFINITY) ? 0.f : expf(p[u] - rowmax);
    lsum += e[u];
  }
  red[t] = lsum; __syncthreads();
  for (int s = 128; s > 0; s >>= 1) {
    if (t < s) red[t] += red[t + s];
    __syncthreads();
  }
  float inv = (red[0] > 0.f) ? 1.f / red[0] : 0.f;
  for (int u = 0; u < 2; ++u) {
    int j = t + u * 256;
    if (j < kN) wgt[i * kN + j] = msk[u] ? e[u] * inv : 0.f;
  }
}

// ---------------------------------------------------------------------------
// Gate kernel: per (i, 64-wide d-block), GEMM r[i,:,:] (384x64) @ Wr (64x64)
// via WMMA; the 48 KB r row-block is DMA'd into LDS by the Tensor Data Mover.
// Fused epilogue: sigmoid(+P1[i,d]+P2[j,d]) * wgt[i,j] * h_j[d], reduced
// over j into hsum[i,d].
// ---------------------------------------------------------------------------
__global__ void k_gate(const _Float16* __restrict__ r16,
                       const _Float16* __restrict__ wrT,
                       const float* __restrict__ P1,
                       const float* __restrict__ P2,
                       const float* __restrict__ wgt,
                       const float* __restrict__ self_h,
                       float* __restrict__ hsum) {
  __shared__ alignas(32) _Float16 rrs[kN][kR];    // 48 KB
  __shared__ alignas(32) _Float16 wrTs[64][kR];   // 8 KB
  __shared__ float wgts[kN];
  __shared__ float p1s[64];
  __shared__ float hpart[64];
  int i = blockIdx.y, db = blockIdx.x;
  int tid = threadIdx.x;
  int wave = tid >> 5, lane = tid & 31;
  int lo = lane & 15, hi = lane >> 4, b8 = hi * 8;

#if HAVE_TDM
  if (wave == 0)
    tdm_load_halves(&rrs[0][0], r16 + (size_t)i * kN * kR, kN * kR);  // 48 KB via TDM
#else
  {
    const uint4* src = (const uint4*)(r16 + (size_t)i * kN * kR);
    uint4* dst = (uint4*)&rrs[0][0];
    for (int idx = tid; idx < 3072; idx += 256) dst[idx] = src[idx];
  }
#endif
  { // Wr^T slice for this d-block: 64*64 halves = 8 KB = 512 uint4
    const uint4* src = (const uint4*)(wrT + db * 64 * kR);
    uint4* dst = (uint4*)&wrTs[0][0];
    for (int idx = tid; idx < 512; idx += 256) dst[idx] = src[idx];
  }
  for (int j = tid; j < kN; j += 256) wgts[j] = wgt[i * kN + j];
  if (tid < 64) { p1s[tid] = P1[i * kD + db * 64 + tid]; hpart[tid] = 0.f; }
  __syncthreads();

  float hacc[4] = {0, 0, 0, 0};
  for (int t3 = 0; t3 < 3; ++t3) {
    int jt = wave + 8 * t3;           // 24 j-tiles over 8 waves
    int row = jt * 16 + lo;
    h16 af[2];
    for (int kc = 0; kc < 2; ++kc) {
      const h8* pa = (const h8*)&rrs[row][kc * 32 + hi * 16];
      h8 alo = pa[0], ahi = pa[1];
      for (int e = 0; e < 8; ++e) { af[kc][e] = alo[e]; af[kc][8 + e] = ahi[e]; }
    }
    for (int nt = 0; nt < 4; ++nt) {
      f8 acc = {};
      for (int kc = 0; kc < 2; ++kc) {
        h16 bf = *(const h16*)&wrTs[nt * 16 + lo][kc * 32 + hi * 16];
        acc = wmma_f16f32(af[kc], bf, acc);
      }
      int dl = nt * 16 + lo;
      int d  = db * 64 + dl;
      float p1b = p1s[dl];
      float part = 0.f;
      for (int v = 0; v < 8; ++v) {
        int j = jt * 16 + b8 + v;     // accumulator row m = b8+v
        float x = acc[v] + p1b + P2[j * kD + d];
        float g = 1.f / (1.f + __expf(-x));
        part = fmaf(wgts[j] * self_h[j * kD + d], g, part);
      }
      hacc[nt] += part;
    }
  }
  for (int nt = 0; nt < 4; ++nt) {
    float s = hacc[nt] + __shfl_xor(hacc[nt], 16, 32);
    if (hi == 0) atomicAdd(&hpart[nt * 16 + lo], s);
  }
  __syncthreads();
  if (tid < 64) hsum[i * kD + db * 64 + tid] = hpart[tid];
}

// ---------------------------------------------------------------------------
// Output MLP + cell/gate math; writes (outgate, H_out, C) concatenated.
// ---------------------------------------------------------------------------
__global__ void k_final(const float* __restrict__ hsum,
                        const float* __restrict__ self_c,
                        const float* __restrict__ outgate,
                        const float* __restrict__ w1, const float* __restrict__ b1,
                        const float* __restrict__ w2, const float* __restrict__ b2,
                        float* __restrict__ out) {
  int i = blockIdx.x, t = threadIdx.x;
  __shared__ float hs[kD];
  __shared__ float hid[kNeiH];
  hs[t] = hsum[i * kD + t];
  __syncthreads();
  float a = 0.f;
  for (int k = 0; k < kD; ++k) a = fmaf(hs[k], w1[k * kNeiH + t], a);
  hid[t] = fmaxf(a + b1[t], 0.f);
  __syncthreads();
  float c = 0.f;
  for (int k = 0; k < kNeiH; ++k) c = fmaf(hid[k], w2[k * kD + t], c);
  c += b2[t] + self_c[i * kD + t];
  float og = outgate[i * kD + t];
  const int ND = kN * kD;
  out[i * kD + t]            = og;
  out[ND + i * kD + t]       = og * tanhf(c);
  out[2 * ND + i * kD + t]   = c;
}

// ---------------------------------------------------------------------------
extern "C" void kernel_launch(void* const* d_in, const int* in_sizes, int n_in,
                              void* d_out, int out_size, void* d_ws, size_t ws_size,
                              hipStream_t stream) {
  const float* corr    = (const float*)d_in[0];
  const int*   nei     = (const int*)d_in[1];
  /* d_in[2] = nei_num (unused by the reference) */
  const float* outg    = (const float*)d_in[3];
  const float* self_h  = (const float*)d_in[4];
  const float* self_c  = (const float*)d_in[5];
  const float* rel_w1  = (const float*)d_in[6];
  const float* rel_b1  = (const float*)d_in[7];
  const float* rel_w2  = (const float*)d_in[8];
  const float* rel_b2  = (const float*)d_in[9];
  const float* ngate_w = (const float*)d_in[10];
  const float* ngate_b = (const float*)d_in[11];
  const float* war_w   = (const float*)d_in[12];
  const float* war_b   = (const float*)d_in[13];
  const float* wnei_w1 = (const float*)d_in[14];
  const float* wnei_b1 = (const float*)d_in[15];
  const float* wnei_w2 = (const float*)d_in[16];
  const float* wnei_b2 = (const float*)d_in[17];

  char* p = (char*)d_ws;
  auto carve = [&](size_t bytes) {
    char* r = p;
    p += (bytes + 255) & ~(size_t)255;
    return (void*)r;
  };
  _Float16* r16 = (_Float16*)carve((size_t)kN * kN * kR * 2);  // 18.9 MB
  float*    tt  = (float*)carve((size_t)kN * kN * 4);
  float*    wgt = (float*)carve((size_t)kN * kN * 4);
  _Float16* wrT = (_Float16*)carve((size_t)kD * kR * 2);
  _Float16* w2T = (_Float16*)carve((size_t)kR * kRelH * 2);
  float*    P1  = (float*)carve((size_t)kN * kD * 4);
  float*    P2  = (float*)carve((size_t)kN * kD * 4);
  float*    q1  = (float*)carve((size_t)kN * 4);
  float*    q2  = (float*)carve((size_t)kN * 4);
  float*    hsm = (float*)carve((size_t)kN * kD * 4);

  k_prep_convert<<<64, 256, 0, stream>>>(ngate_w, rel_w2, wrT, w2T);
  k_prep_rows<<<kN, 256, 0, stream>>>(self_h, ngate_w, ngate_b, war_w, war_b,
                                      P1, P2, q1, q2);
  k_pairs<<<(kN * kN) / 128, 256, 0, stream>>>(corr, rel_w1, rel_b1, rel_b2,
                                               w2T, war_w, q1, q2, r16, tt);
  k_softmax<<<kN, 256, 0, stream>>>(tt, nei, wgt);
  k_gate<<<dim3(4, kN), 256, 0, stream>>>(r16, wrT, P1, P2, wgt, self_h, hsm);
  k_final<<<kN, 256, 0, stream>>>(hsm, self_c, outg,
                                  wnei_w1, wnei_b1, wnei_w2, wnei_b2,
                                  (float*)d_out);
}
